// DiscretisedRKHSEmbedding_35356170781152
// MI455X (gfx1250) — compile-verified, matrix operand order checked by reading, not run on hardware
//
#include <hip/hip_runtime.h>

typedef __attribute__((ext_vector_type(16))) _Float16 v16h;
typedef __attribute__((ext_vector_type(8)))  _Float16 v8h;
typedef __attribute__((ext_vector_type(8)))  float    v8f;

#define NB     16
#define NPTS   1024
#define NGRID  4096
#define CY     8      // raw Y channels
#define CB     9      // Yb channels (1 + CY)

#define LOG2E  1.4426950408889634f

// ---------------------------------------------------------------- grid output
__global__ void grid_out_kernel(float* __restrict__ out) {
    int i = blockIdx.x * blockDim.x + threadIdx.x;
    if (i < NGRID) {
        const float step = 6.0f / 63.0f;
        out[2 * i + 0] = -3.0f + step * (float)(i >> 6);
        out[2 * i + 1] = -3.0f + step * (float)(i & 63);
    }
}

// ---------------------------------------------------------------- main kernel
// block = 128 threads = 4 waves; each wave owns one 16-row m-tile.
// gridDim.x = NB * 64  (64 blocks x 4 waves = 256 m-tiles of 16 rows = 4096)
__global__ __launch_bounds__(128)
void rkhs_wmma_kernel(const float* __restrict__ X,
                      const float* __restrict__ Y,
                      float* __restrict__ out) {
    __shared__ float    sXp[NPTS * 2];       // x0,x1 pairs           (8 KB)
    __shared__ float    sX2s[NPTS];          // -2*log2e*|x|^2        (4 KB)
    __shared__ _Float16 sYbT[16 * NPTS];     // padded Yb, [c][n]     (32 KB)

    const int tid  = threadIdx.x;
    const int wave = tid >> 5;
    const int lane = tid & 31;
    const int b    = blockIdx.x >> 6;                 // batch
    const int mt   = (blockIdx.x & 63) * 4 + wave;    // m-tile 0..255

    const float* Xg = X + (size_t)b * NPTS * 2;
    const float* Yg = Y + (size_t)b * NPTS * CY;

    // ---- stage batch data into LDS (once per block) ----
    for (int n = tid; n < NPTS; n += 128) {
        float x0 = Xg[2 * n + 0];
        float x1 = Xg[2 * n + 1];
        sXp[2 * n + 0] = x0;
        sXp[2 * n + 1] = x1;
        sX2s[n] = (-2.0f * LOG2E) * (x0 * x0 + x1 * x1);
    }
    // transposed, zero-padded, f16 Yb table: sYbT[c*NPTS + n]
    for (int i = tid; i < 16 * NPTS; i += 128) {
        const int cc = i >> 10;          // 0..15
        const int nn = i & (NPTS - 1);
        float v;
        if (cc == 0)       v = 1.0f;
        else if (cc <= CY) v = Yg[nn * CY + (cc - 1)];
        else               v = 0.0f;
        sYbT[i] = (_Float16)v;
    }
    __syncthreads();

    // ---- per-lane geometry ----
    const int mloc = lane & 15;          // A row / B column / output channel
    const int half = lane >> 4;          // lane-half selects k sub-ranges
    const int m    = mt * 16 + mloc;
    const float step = 6.0f / 63.0f;
    const float gx  = -3.0f + step * (float)(m >> 6);
    const float gy  = -3.0f + step * (float)(m & 63);
    // K = exp(-2*d2) = exp2( g2s + sX2s[n] + gxs*x0 + gys*x1 )
    const float g2s = (-2.0f * LOG2E) * (gx * gx + gy * gy);
    const float gxs = (4.0f * LOG2E) * gx;
    const float gys = (4.0f * LOG2E) * gy;
    const int   c   = mloc;              // Yb channel this lane supplies/stores

    const _Float16* rowB = sYbT + c * NPTS;

    v8f acc = {};

    // ---- K-accumulation: 32 steps of k=32 ----
    for (int n0 = 0; n0 < NPTS; n0 += 32) {
        // B fragment: two contiguous 8-half (16B, aligned) LDS loads
        const v8h blo = *(const v8h*)(rowB + n0 + 8 * half);
        const v8h bhi = *(const v8h*)(rowB + n0 + 16 + 8 * half);
        v16h bfrag;
#pragma unroll
        for (int e = 0; e < 8; ++e) { bfrag[e] = blo[e]; bfrag[8 + e] = bhi[e]; }

        // A fragment: 16 exp2 evaluations in WMMA A layout
        v16h afrag;
#pragma unroll
        for (int e = 0; e < 16; ++e) {
            const int n = n0 + (e & 7) + ((e >> 3) << 4) + (half << 3);
            const float x0  = sXp[2 * n + 0];
            const float x1  = sXp[2 * n + 1];
            const float arg = fmaf(gxs, x0, fmaf(gys, x1, g2s + sX2s[n]));
            afrag[e] = (_Float16)__builtin_amdgcn_exp2f(arg);
        }

        acc = __builtin_amdgcn_wmma_f32_16x16x32_f16(
                  false, afrag, false, bfrag, (short)0, acc, false, false);
    }

    // ---- epilogue: density normalization + store ----
    // C/D layout: lane holds column N = lane&15, rows M = r + 8*half (r = 0..7)
    float* Og = out + 2 * NGRID + (size_t)b * NGRID * CB;
    const int srcLane = half << 4;       // lane holding N==0 for this half
#pragma unroll
    for (int r = 0; r < 8; ++r) {
        const float v    = acc[r];
        const float dens = __shfl(v, srcLane, 32);
        // fast reciprocal (v_rcp_f32) instead of IEEE divide expansion
        const float rinv = __builtin_amdgcn_rcpf(dens + 1e-6f);
        if (c < CB) {
            const int mrow = mt * 16 + r + 8 * half;
            const float o  = (c == 0) ? dens : v * rinv;
            Og[(size_t)mrow * CB + c] = o;
        }
    }
}

// ---------------------------------------------------------------- launcher
extern "C" void kernel_launch(void* const* d_in, const int* in_sizes, int n_in,
                              void* d_out, int out_size, void* d_ws, size_t ws_size,
                              hipStream_t stream) {
    const float* X = (const float*)d_in[0];   // (16,1024,2) f32
    const float* Y = (const float*)d_in[1];   // (16,1024,8) f32
    float* out = (float*)d_out;               // 8192 grid + 16*4096*9 Y_grid

    grid_out_kernel<<<(NGRID + 255) / 256, 256, 0, stream>>>(out);
    rkhs_wmma_kernel<<<NB * 64, 128, 0, stream>>>(X, Y, out);
}
